// DS_RNN_II_11690900980420
// MI455X (gfx1250) — matrix operand order repaired and checked by reading, not compiled, not run on hardware
//
#include <hip/hip_runtime.h>
#include <math.h>

// ---------------- dims (fixed by the reference) ----------------
#define Bz   512
#define TEnc 96
#define TDec 24
#define In   64
#define Hd   512

// ---------------- WMMA / vector types ----------------
typedef __attribute__((ext_vector_type(16))) __bf16         v16bf;
typedef __attribute__((ext_vector_type(8)))  float          v8f;
typedef __attribute__((ext_vector_type(8)))  unsigned short ushort8;
typedef __attribute__((__vector_size__(4 * sizeof(int)))) int v4i;

union Frag { ushort8 u[2]; v16bf v; };

#define ASG __attribute__((address_space(1)))
#define ASL __attribute__((address_space(3)))

#if __has_builtin(__builtin_amdgcn_global_load_async_to_lds_b128)
#define HAVE_ASYNC_COPY 1
#else
#define HAVE_ASYNC_COPY 0
#endif

__device__ __forceinline__ unsigned short f2bf(float f) {
    unsigned int u = __float_as_uint(f);
    u += 0x7FFFu + ((u >> 16) & 1u);   // round-to-nearest-even
    return (unsigned short)(u >> 16);
}
__device__ __forceinline__ float sigf(float x) { return 1.0f / (1.0f + expf(-x)); }

// ---------------- generic tiled WMMA GEMM: C[M,N] (+)= A[M,K] @ W[N,K]^T (+bias) ----------------
// A: f32 (converted to bf16 during LDS staging, optional fused tanh)
// W: bf16 (pre-converted once), staged via async global->LDS DMA when available
#define BM 128
#define BN 128
#define BK 32
#define LDP (BK + 8)   // 40 ushorts/row -> 80B row stride, 16B aligned

__global__ __launch_bounds__(256)
void wmma_gemm_kernel(const float* __restrict__ A, int lda,
                      const unsigned short* __restrict__ W, int ldw,
                      const float* __restrict__ bias,
                      float* __restrict__ C, int ldc,
                      int M, int N, int K, int accumulate, int tanhA)
{
    __shared__ unsigned short As[BM][LDP];
    __shared__ unsigned short Bs[BN][LDP];

    const int tid   = threadIdx.x;
    const int lane  = tid & 31;
    const int wave  = tid >> 5;       // 0..7
    const int wm    = wave & 1;       // 2 waves along M (64 rows each)
    const int wn    = wave >> 1;      // 4 waves along N (32 cols each)
    const int tileM = blockIdx.y * BM;
    const int tileN = blockIdx.x * BN;

    const int lrow = tid >> 3;        // 0..31 : staged row
    const int lcol = (tid & 7) * 4;   // 0..28 : staged K offset (float4)

    v8f acc[4][2];
    v8f zv = {};
#pragma unroll
    for (int i = 0; i < 4; ++i)
#pragma unroll
        for (int j = 0; j < 2; ++j) acc[i][j] = zv;

    const int l15  = lane & 15;
    const int hsel = lane >> 4;

    for (int k0 = 0; k0 < K; k0 += BK) {
        // ---- stage A tile (f32 -> bf16, optional fused tanh); OOB rows feed only guarded outputs
#pragma unroll
        for (int rr = 0; rr < BM; rr += 32) {
            int r  = lrow + rr;
            int gr = tileM + r;
            if (gr < M) {
                float4 v = *(const float4*)(A + (size_t)gr * lda + k0 + lcol);
                if (tanhA) { v.x = tanhf(v.x); v.y = tanhf(v.y); v.z = tanhf(v.z); v.w = tanhf(v.w); }
                As[r][lcol + 0] = f2bf(v.x);
                As[r][lcol + 1] = f2bf(v.y);
                As[r][lcol + 2] = f2bf(v.z);
                As[r][lcol + 3] = f2bf(v.w);
            }
        }
        // ---- stage B tile (bf16 weights): async global->LDS DMA, 16B per lane, 2 per thread ----
#pragma unroll
        for (int i = 0; i < 2; ++i) {
            int idx = tid + i * 256;
            int r   = idx >> 2;
            int ch  = (idx & 3) * 8;      // ushort offset of the 16B chunk within the row
            int gn  = tileN + r;
            if (gn < N) {
                const unsigned short* gp = W + (size_t)gn * ldw + k0 + ch;
#if HAVE_ASYNC_COPY
                __builtin_amdgcn_global_load_async_to_lds_b128(
                    (ASG v4i*)(uintptr_t)gp,
                    (ASL v4i*)(unsigned)(uintptr_t)&Bs[r][ch],
                    0, 0);
#else
                *(ushort8*)&Bs[r][ch] = *(const ushort8*)gp;
#endif
            }
        }
#if HAVE_ASYNC_COPY
#if __has_builtin(__builtin_amdgcn_s_wait_asynccnt)
        __builtin_amdgcn_s_wait_asynccnt(0);
#else
        asm volatile("s_wait_asynccnt 0x0" ::: "memory");
#endif
#endif
        __syncthreads();

        // speculative prefetch of next A K-tile (gfx1250 global_prefetch_b8)
        if (k0 + BK < K) {
            int gr = tileM + lrow;
            if (gr < M) __builtin_prefetch(A + (size_t)gr * lda + k0 + BK + lcol, 0, 3);
        }

        // ---- fragments from LDS (ISA 16-bit A 16x32 / B 32x16 layouts) ----
        Frag a[4], b[2];
#pragma unroll
        for (int mi = 0; mi < 4; ++mi) {
            int r  = wm * 64 + mi * 16 + l15;
            int kb = hsel * 8;
            a[mi].u[0] = *(const ushort8*)&As[r][kb];
            a[mi].u[1] = *(const ushort8*)&As[r][kb + 16];
        }
#pragma unroll
        for (int ni = 0; ni < 2; ++ni) {
            int n  = wn * 32 + ni * 16 + l15;
            int kb = hsel * 8;
            b[ni].u[0] = *(const ushort8*)&Bs[n][kb];
            b[ni].u[1] = *(const ushort8*)&Bs[n][kb + 16];
        }
#pragma unroll
        for (int mi = 0; mi < 4; ++mi)
#pragma unroll
            for (int ni = 0; ni < 2; ++ni)
                acc[mi][ni] = __builtin_amdgcn_wmma_f32_16x16x32_bf16(
                    false, a[mi].v, false, b[ni].v, (short)0, acc[mi][ni], false, false);
        __syncthreads();
    }

    // ---- store: C/D layout -> lane l, vgpr r : row = r + 8*(l>=16), col = l&15 ----
#pragma unroll
    for (int mi = 0; mi < 4; ++mi)
#pragma unroll
        for (int ni = 0; ni < 2; ++ni) {
            int col   = tileN + wn * 32 + ni * 16 + l15;
            int rbase = tileM + wm * 64 + mi * 16 + hsel * 8;
            if (col < N) {
                float bv = bias ? bias[col] : 0.0f;
#pragma unroll
                for (int r = 0; r < 8; ++r) {
                    int row = rbase + r;
                    if (row < M) {
                        size_t off = (size_t)row * ldc + col;
                        float v = acc[mi][ni][r] + bv;
                        if (accumulate) v += C[off];
                        C[off] = v;
                    }
                }
            }
        }
}

// ---------------- small kernels ----------------
__global__ void zero_kernel(float* p, int n) {
    int i = blockIdx.x * blockDim.x + threadIdx.x;
    if (i < n) p[i] = 0.0f;
}

__global__ void add_vec_kernel(const float* a, const float* b, float* o, int n) {
    int i = blockIdx.x * blockDim.x + threadIdx.x;
    if (i < n) o[i] = a[i] + b[i];
}

__global__ void cvt_bf16_kernel(const float* __restrict__ s, unsigned short* __restrict__ d, int n) {
    int i = blockIdx.x * blockDim.x + threadIdx.x;
    if (i < n) d[i] = f2bf(s[i]);
}

// one wave32 per batch row: softmax over I=64 scores, xi = x * alpha
__global__ void softmax_xi_kernel(const float* __restrict__ S, const float* __restrict__ inputs,
                                  int t, float* __restrict__ xi)
{
    int gw   = (blockIdx.x * blockDim.x + threadIdx.x) >> 5;
    int lane = threadIdx.x & 31;
    if (gw >= Bz) return;
    int b = gw;
    float v0 = S[(size_t)b * In + lane];
    float v1 = S[(size_t)b * In + 32 + lane];
    float m = fmaxf(v0, v1);
    for (int off = 16; off; off >>= 1) m = fmaxf(m, __shfl_xor(m, off, 32));
    float e0 = expf(v0 - m), e1 = expf(v1 - m);
    float s = e0 + e1;
    for (int off = 16; off; off >>= 1) s += __shfl_xor(s, off, 32);
    float inv = 1.0f / s;
    const float* x = inputs + ((size_t)b * TEnc + t) * In;
    xi[(size_t)b * In + lane]      = x[lane]      * e0 * inv;
    xi[(size_t)b * In + 32 + lane] = x[32 + lane] * e1 * inv;
}

// gates [B,4H] (i,f,g,o) -> update packed hc [B,2H] ; optionally emit h into hout [B,H]
__global__ void lstm_update_kernel(const float* __restrict__ g, float* __restrict__ hc,
                                   float* __restrict__ hout)
{
    int idx = blockIdx.x * blockDim.x + threadIdx.x;
    if (idx >= Bz * Hd) return;
    int b = idx >> 9, h = idx & (Hd - 1);
    const float* gb = g + (size_t)b * 4 * Hd;
    float ig = sigf(gb[h]);
    float fg = sigf(gb[Hd + h]);
    float gg = tanhf(gb[2 * Hd + h]);
    float og = sigf(gb[3 * Hd + h]);
    float c  = fg * hc[(size_t)b * 2 * Hd + Hd + h] + ig * gg;
    float hh = og * tanhf(c);
    hc[(size_t)b * 2 * Hd + h]      = hh;
    hc[(size_t)b * 2 * Hd + Hd + h] = c;
    if (hout) hout[(size_t)b * Hd + h] = hh;
}

// one wave32 per (b,t): s[b,t] = sum_h tanh(q[b,h]+wx[t,b,h]) * Vw[h] + Vb
__global__ void dec_scores_kernel(const float* __restrict__ q, const float* __restrict__ wx,
                                  const float* __restrict__ Vw, const float* __restrict__ Vb,
                                  float* __restrict__ sbuf)
{
    int gw   = (blockIdx.x * blockDim.x + threadIdx.x) >> 5;
    int lane = threadIdx.x & 31;
    if (gw >= Bz * TEnc) return;
    int b = gw / TEnc, t = gw % TEnc;
    const float* qb  = q  + (size_t)b * Hd;
    const float* wxp = wx + ((size_t)t * Bz + b) * Hd;
    float sum = 0.0f;
    for (int h = lane; h < Hd; h += 32) sum += tanhf(qb[h] + wxp[h]) * Vw[h];
    for (int off = 16; off; off >>= 1) sum += __shfl_xor(sum, off, 32);
    if (lane == 0) sbuf[(size_t)b * TEnc + t] = sum + Vb[0];
}

// din[b,h] = sum_t s[b,t] * mid[t,b,h]
__global__ void dec_context_kernel(const float* __restrict__ sbuf, const float* __restrict__ mid,
                                   float* __restrict__ din)
{
    int idx = blockIdx.x * blockDim.x + threadIdx.x;
    if (idx >= Bz * Hd) return;
    int b = idx >> 9, h = idx & (Hd - 1);
    float sum = 0.0f;
    for (int t = 0; t < TEnc; ++t)
        sum += sbuf[(size_t)b * TEnc + t] * mid[((size_t)t * Bz + b) * Hd + h];
    din[(size_t)b * Hd + h] = sum;
}

// one wave32 per batch row: out[b, j] = dot(h_dec[b], reg_w) + reg_b
__global__ void regress_kernel(const float* __restrict__ hc_dec, const float* __restrict__ rw,
                               const float* __restrict__ rb, float* __restrict__ out, int j)
{
    int gw   = (blockIdx.x * blockDim.x + threadIdx.x) >> 5;
    int lane = threadIdx.x & 31;
    if (gw >= Bz) return;
    int b = gw;
    float sum = 0.0f;
    for (int h = lane; h < Hd; h += 32) sum += hc_dec[(size_t)b * 2 * Hd + h] * rw[h];
    for (int off = 16; off; off >>= 1) sum += __shfl_xor(sum, off, 32);
    if (lane == 0) out[(size_t)b * TDec + j] = sum + rb[0];
}

// ---------------- host-side helpers ----------------
static inline void gemm(hipStream_t s, const float* A, int lda, const unsigned short* W, int ldw,
                        const float* bias, float* C, int ldc, int M, int N, int K,
                        int accumulate, int tanhA)
{
    dim3 grid((N + BN - 1) / BN, (M + BM - 1) / BM), blk(256);
    wmma_gemm_kernel<<<grid, blk, 0, s>>>(A, lda, W, ldw, bias, C, ldc, M, N, K, accumulate, tanhA);
}

extern "C" void kernel_launch(void* const* d_in, const int* in_sizes, int n_in,
                              void* d_out, int out_size, void* d_ws, size_t ws_size,
                              hipStream_t stream)
{
    const float* inputs  = (const float*)d_in[0];
    // d_in[1] = label_p (unused by the reference computation)
    const float* enc_Wih = (const float*)d_in[2];
    const float* enc_Whh = (const float*)d_in[3];
    const float* enc_bih = (const float*)d_in[4];
    const float* enc_bhh = (const float*)d_in[5];
    const float* dec_Wih = (const float*)d_in[6];
    const float* dec_Whh = (const float*)d_in[7];
    const float* dec_bih = (const float*)d_in[8];
    const float* dec_bhh = (const float*)d_in[9];
    const float* Wi_w    = (const float*)d_in[10];
    const float* Wi_b    = (const float*)d_in[11];
    const float* We_w    = (const float*)d_in[12];
    const float* Vd_w    = (const float*)d_in[13];
    const float* Vd_b    = (const float*)d_in[14];
    const float* Wx_w    = (const float*)d_in[15];
    const float* Wx_b    = (const float*)d_in[16];
    const float* Wh_w    = (const float*)d_in[17];
    const float* V_w     = (const float*)d_in[18];
    const float* V_b     = (const float*)d_in[19];
    const float* mid_Wih = (const float*)d_in[20];
    const float* mid_Whh = (const float*)d_in[21];
    const float* mid_bih = (const float*)d_in[22];
    const float* mid_bhh = (const float*)d_in[23];
    const float* reg_w   = (const float*)d_in[24];
    const float* reg_b   = (const float*)d_in[25];
    float* out = (float*)d_out;

    // ---- workspace carve-out: f32 region ----
    float* w = (float*)d_ws;
    size_t o = 0;
    float* mid   = w + o; o += (size_t)TEnc * Bz * Hd;   // [T,B,H]
    float* wx    = w + o; o += (size_t)TEnc * Bz * Hd;   // [T,B,H]
    float* P     = w + o; o += (size_t)Bz * Hd;          // spatial attn pre-tanh
    float* S     = w + o; o += (size_t)Bz * In;          // spatial scores
    float* xi    = w + o; o += (size_t)Bz * In;
    float* G     = w + o; o += (size_t)Bz * 4 * Hd;      // gates
    float* hc_e  = w + o; o += (size_t)Bz * 2 * Hd;      // packed [h|c]
    float* hc_m  = w + o; o += (size_t)Bz * 2 * Hd;
    float* hc_d  = w + o; o += (size_t)Bz * 2 * Hd;
    float* qb    = w + o; o += (size_t)Bz * Hd;
    float* sbuf  = w + o; o += (size_t)Bz * TEnc;
    float* din   = w + o; o += (size_t)Bz * Hd;
    float* bse   = w + o; o += 4 * Hd;
    float* bsm   = w + o; o += 4 * Hd;
    float* bsd   = w + o; o += 4 * Hd;
    // ---- bf16 weight region ----
    unsigned short* u = (unsigned short*)(w + o);
    size_t uo = 0;
    unsigned short* bWi    = u + uo; uo += (size_t)Hd * In;
    unsigned short* bWe    = u + uo; uo += (size_t)Hd * 2 * Hd;
    unsigned short* bVd    = u + uo; uo += (size_t)In * Hd;
    unsigned short* bEncIh = u + uo; uo += (size_t)4 * Hd * In;
    unsigned short* bEncHh = u + uo; uo += (size_t)4 * Hd * Hd;
    unsigned short* bMidIh = u + uo; uo += (size_t)4 * Hd * Hd;
    unsigned short* bMidHh = u + uo; uo += (size_t)4 * Hd * Hd;
    unsigned short* bWx    = u + uo; uo += (size_t)Hd * Hd;
    unsigned short* bWh    = u + uo; uo += (size_t)Hd * 2 * Hd;
    unsigned short* bDecIh = u + uo; uo += (size_t)4 * Hd * Hd;
    unsigned short* bDecHh = u + uo; uo += (size_t)4 * Hd * Hd;
    (void)ws_size; (void)in_sizes; (void)n_in; (void)out_size;

    // zero the three packed hc state buffers (contiguous)
    {
        int n = 3 * Bz * 2 * Hd;
        zero_kernel<<<(n + 255) / 256, 256, 0, stream>>>(hc_e, n);
    }
    // combined LSTM biases (bih + bhh)
    add_vec_kernel<<<(4 * Hd + 255) / 256, 256, 0, stream>>>(enc_bih, enc_bhh, bse, 4 * Hd);
    add_vec_kernel<<<(4 * Hd + 255) / 256, 256, 0, stream>>>(mid_bih, mid_bhh, bsm, 4 * Hd);
    add_vec_kernel<<<(4 * Hd + 255) / 256, 256, 0, stream>>>(dec_bih, dec_bhh, bsd, 4 * Hd);
    // one-time f32 -> bf16 weight conversion (stays L2-resident across all recurrent steps)
    {
        auto cvt = [&](const float* s, unsigned short* d, int n) {
            cvt_bf16_kernel<<<(n + 255) / 256, 256, 0, stream>>>(s, d, n);
        };
        cvt(Wi_w,    bWi,    Hd * In);
        cvt(We_w,    bWe,    Hd * 2 * Hd);
        cvt(Vd_w,    bVd,    In * Hd);
        cvt(enc_Wih, bEncIh, 4 * Hd * In);
        cvt(enc_Whh, bEncHh, 4 * Hd * Hd);
        cvt(mid_Wih, bMidIh, 4 * Hd * Hd);
        cvt(mid_Whh, bMidHh, 4 * Hd * Hd);
        cvt(Wx_w,    bWx,    Hd * Hd);
        cvt(Wh_w,    bWh,    Hd * 2 * Hd);
        cvt(dec_Wih, bDecIh, 4 * Hd * Hd);
        cvt(dec_Whh, bDecHh, 4 * Hd * Hd);
    }

    // ---- fused encoder + mid-LSTM recurrence over T_ENC ----
    for (int t = 0; t < TEnc; ++t) {
        // P = x_t @ Wi^T + Wi_b ; P += [h,c] @ We^T
        gemm(stream, inputs + (size_t)t * In, TEnc * In, bWi, In, Wi_b, P, Hd, Bz, Hd, In, 0, 0);
        gemm(stream, hc_e, 2 * Hd, bWe, 2 * Hd, nullptr, P, Hd, Bz, Hd, 2 * Hd, 1, 0);
        // S = tanh(P) @ Vd^T + Vd_b
        gemm(stream, P, Hd, bVd, Hd, Vd_b, S, In, Bz, In, Hd, 0, 1);
        softmax_xi_kernel<<<(Bz * 32 + 255) / 256, 256, 0, stream>>>(S, inputs, t, xi);
        // gates = xi @ Wih^T + (bih+bhh) ; gates += h @ Whh^T
        gemm(stream, xi, In, bEncIh, In, bse, G, 4 * Hd, Bz, 4 * Hd, In, 0, 0);
        gemm(stream, hc_e, 2 * Hd, bEncHh, Hd, nullptr, G, 4 * Hd, Bz, 4 * Hd, Hd, 1, 0);
        lstm_update_kernel<<<(Bz * Hd + 255) / 256, 256, 0, stream>>>(G, hc_e, nullptr);
        // mid LSTM consumes the fresh encoder h
        gemm(stream, hc_e, 2 * Hd, bMidIh, Hd, bsm, G, 4 * Hd, Bz, 4 * Hd, Hd, 0, 0);
        gemm(stream, hc_m, 2 * Hd, bMidHh, Hd, nullptr, G, 4 * Hd, Bz, 4 * Hd, Hd, 1, 0);
        lstm_update_kernel<<<(Bz * Hd + 255) / 256, 256, 0, stream>>>(G, hc_m, mid + (size_t)t * Bz * Hd);
    }

    // ---- one big parallel GEMM: wx_mid = mid @ Wx^T + Wx_b  (49152 x 512 x 512) ----
    gemm(stream, mid, Hd, bWx, Hd, Wx_b, wx, Hd, TEnc * Bz, Hd, Hd, 0, 0);

    // ---- temporal-attention decoder (T_DEC + 4 steps) ----
    for (int t = 0; t < TDec + 4; ++t) {
        gemm(stream, hc_d, 2 * Hd, bWh, 2 * Hd, nullptr, qb, Hd, Bz, Hd, 2 * Hd, 0, 0);
        dec_scores_kernel<<<(Bz * TEnc * 32 + 255) / 256, 256, 0, stream>>>(qb, wx, V_w, V_b, sbuf);
        dec_context_kernel<<<(Bz * Hd + 255) / 256, 256, 0, stream>>>(sbuf, mid, din);
        gemm(stream, din, Hd, bDecIh, Hd, bsd, G, 4 * Hd, Bz, 4 * Hd, Hd, 0, 0);
        gemm(stream, hc_d, 2 * Hd, bDecHh, Hd, nullptr, G, 4 * Hd, Bz, 4 * Hd, Hd, 1, 0);
        lstm_update_kernel<<<(Bz * Hd + 255) / 256, 256, 0, stream>>>(G, hc_d, nullptr);
        if (t >= 4)
            regress_kernel<<<(Bz * 32 + 255) / 256, 256, 0, stream>>>(hc_d, reg_w, reg_b, out, t - 4);
    }
}
